// BertNetModel_76218489634925
// MI455X (gfx1250) — matrix operand order repaired and checked by reading, not compiled, not run on hardware
//
#include <hip/hip_runtime.h>
#include <hip/hip_bf16.h>
#include <math.h>

#define DEVINL __device__ __forceinline__

typedef __attribute__((ext_vector_type(16))) __bf16 v16bf;
typedef __attribute__((ext_vector_type(8)))  float  v8f;
typedef __attribute__((ext_vector_type(4)))  unsigned u32x4;
typedef __attribute__((ext_vector_type(8)))  int     i32x8;
typedef __attribute__((ext_vector_type(4)))  int     i32x4;

// ---------------- problem sizes ----------------
constexpr int B_ = 8, T_ = 256, K_ = 20, H_ = 768, V_ = 30522;
constexpr int BT   = B_ * T_;        // 2048 tokens
constexpr int TM1R = (T_ - 1) * B_;  // 2040 LSTM rows (time-major r = t*8+b)
constexpr int G4H  = 4 * H_;         // 3072 gates
constexpr int VPAD = 30592;          // 239 chunks * 128 cols
constexpr int NCHUNK = 239;
constexpr int AP = H_ + 8;           // LDS row pitch (bf16): 1552 B, 16B aligned
constexpr float TAU = 1.0f;
constexpr float Z_BETA = 1.0f;

// ---------------- workspace arena ----------------
constexpr size_t alignUp(size_t x) { return (x + 255) & ~size_t(255); }
constexpr size_t OFF_TRANS = 0;
constexpr size_t OFF_EMI   = alignUp(OFF_TRANS + (size_t)K_ * K_ * 4);
constexpr size_t OFF_ALPHA = alignUp(OFF_EMI   + (size_t)BT * K_ * 4);
constexpr size_t OFF_RELAX = alignUp(OFF_ALPHA + (size_t)T_ * B_ * K_ * 4);
constexpr size_t OFF_ZIDS  = alignUp(OFF_RELAX + (size_t)T_ * B_ * K_ * 4);
constexpr size_t OFF_SCAL  = alignUp(OFF_ZIDS  + (size_t)T_ * B_ * 4);
constexpr size_t OFF_C     = alignUp(OFF_SCAL  + 64);
constexpr size_t OFF_HBF   = alignUp(OFF_C     + (size_t)B_ * H_ * 4);
constexpr size_t OFF_GATES = alignUp(OFF_HBF   + (size_t)16 * H_ * 2);
constexpr size_t OFF_TGTL  = alignUp(OFF_GATES + (size_t)B_ * G4H * 4);
constexpr size_t OFF_DECIN = alignUp(OFF_TGTL  + (size_t)2048 * 4);
constexpr size_t OFF_HS    = alignUp(OFF_DECIN + (size_t)2048 * H_ * 2);
constexpr size_t OFF_WIH   = alignUp(OFF_HS    + (size_t)2048 * H_ * 2);
constexpr size_t OFF_WHH   = alignUp(OFF_WIH   + (size_t)G4H * H_ * 2);
constexpr size_t OFF_XW    = alignUp(OFF_WHH   + (size_t)G4H * H_ * 2);
constexpr size_t OFF_WOUT  = alignUp(OFF_XW    + (size_t)2048 * G4H * 4);
constexpr size_t OFF_STATS = alignUp(OFF_WOUT  + (size_t)VPAD * H_ * 2);
// total ~92.5 MB

// ---------------- helpers ----------------
DEVINL __bf16 f2b(float f) {  // RNE fp32 -> bf16
  unsigned u = __builtin_bit_cast(unsigned, f);
  unsigned r = (u + 0x7FFFu + ((u >> 16) & 1u)) >> 16;
  return __builtin_bit_cast(__bf16, (unsigned short)r);
}
DEVINL __bf16 zbf() { return __builtin_bit_cast(__bf16, (unsigned short)0); }

DEVINL v8f wmma_bb(v16bf a, v16bf b, v8f c) {
  return __builtin_amdgcn_wmma_f32_16x16x32_bf16(false, a, false, b,
                                                 (short)0, c, false, false);
}

struct U32x8 { uint4 a, b; };  // 32B block == one v16bf operand

// A operand (16x32 bf16, ISA 7.12.2): lane row = lane&15; lanes 0-15 carry
// K {0..7,16..23}, lanes 16-31 carry K {8..15,24..31}. Both 8-element chunks
// are 16B-aligned, so load as two b128s and bit_cast.
DEVINL v16bf load_a(const __bf16* Arow, int k0, int half) {
  U32x8 u;
  u.a = *(const uint4*)(Arow + k0 + half * 8);
  u.b = *(const uint4*)(Arow + k0 + 16 + half * 8);
  return __builtin_bit_cast(v16bf, u);
}
// B operand (32x16 bf16): lane col = lane&15; same K order as A: lanes 0-15
// carry K {0..7,16..23}, lanes 16-31 carry K {8..15,24..31}.
DEVINL v16bf load_b(const __bf16* Brow, int k0, int half) {
  const __bf16* p = Brow + k0;
  U32x8 u;
  u.a = *(const uint4*)(p + half * 8);
  u.b = *(const uint4*)(p + 16 + half * 8);
  return __builtin_bit_cast(v16bf, u);
}

// One wave computes a 16x16 f32 tile over K = H_ = 768 in 32-wide slabs.
DEVINL void wmma_k_loop(const __bf16* __restrict__ Abase, int astride,
                        const __bf16* __restrict__ Brow, v8f& acc) {
  const int lane = threadIdx.x & 31;
  const int half = lane >> 4;
  const int lm   = lane & 15;
  const __bf16* Arow = Abase + (size_t)lm * astride;
  for (int k0 = 0; k0 < H_; k0 += 32) {
    __builtin_prefetch(Brow + k0 + 64, 0, 0);  // global_prefetch_b8
    acc = wmma_bb(load_a(Arow, k0, half), load_b(Brow, k0, half), acc);
  }
}

// ---------------- kernels ----------------

// Convert weights to bf16 once per call; zero the recurrent state & scalars.
__global__ void k_convert(const float* Wih, const float* Whh, const float* Wout,
                          char* ws) {
  __bf16* wih  = (__bf16*)(ws + OFF_WIH);
  __bf16* whh  = (__bf16*)(ws + OFF_WHH);
  __bf16* wout = (__bf16*)(ws + OFF_WOUT);
  __bf16* hbf  = (__bf16*)(ws + OFF_HBF);
  float*  cbuf = (float*)(ws + OFF_C);
  float*  scal = (float*)(ws + OFF_SCAL);
  size_t i = (size_t)blockIdx.x * 256 + threadIdx.x;
  const size_t n0 = (size_t)G4H * H_;
  const size_t n1 = n0 + n0;
  const size_t n2 = n1 + (size_t)V_ * H_;
  const size_t n3 = n2 + (size_t)(VPAD - V_) * H_;
  const size_t n4 = n3 + (size_t)16 * H_;
  const size_t n5 = n4 + (size_t)B_ * H_;
  const size_t n6 = n5 + 16;
  if (i < n0)       wih[i] = f2b(Wih[i]);
  else if (i < n1)  whh[i - n0] = f2b(Whh[i - n0]);
  else if (i < n2)  wout[i - n1] = f2b(Wout[i - n1]);
  else if (i < n3)  wout[(size_t)V_ * H_ + (i - n2)] = zbf();
  else if (i < n4)  hbf[i - n3] = zbf();
  else if (i < n5)  cbuf[i - n4] = 0.f;
  else if (i < n6)  scal[i - n5] = 0.f;
}

// transition[K,K] = S @ S^T  (tiny)
__global__ void k_transition(const float* S, char* ws) {
  float* tr = (float*)(ws + OFF_TRANS);
  int tid = threadIdx.x;
  if (tid < K_ * K_) {
    int i = tid / K_, j = tid % K_;
    float s = 0.f;
    for (int h = 0; h < H_; ++h) s += S[i * H_ + h] * S[j * H_ + h];
    tr[tid] = s;
  }
}

// emission[b*T+t, k] = x_emb row . S row   (WMMA, fp32->bf16 on load)
__global__ void k_emission(const float* xemb, const float* S, char* ws) {
  float* emi = (float*)(ws + OFF_EMI);
  int w = threadIdx.x >> 5;
  int id = blockIdx.x * 8 + w;               // 256 tiles = 128 M x 2 N
  int mt = id >> 1, nt = id & 1;
  int lane = threadIdx.x & 31, half = lane >> 4, lm = lane & 15;
  int n = nt * 16 + lm;
  int nrow = (n < K_) ? n : 0;               // clamp; garbage cols never stored
  const float* Arow = xemb + (size_t)(mt * 16 + lm) * H_;
  const float* Brow = S + (size_t)nrow * H_;
  v8f acc = {};
  for (int k0 = 0; k0 < H_; k0 += 32) {
    v16bf a, b;
#pragma unroll
    for (int j = 0; j < 8; ++j) {
      a[j]     = f2b(Arow[k0 + half * 8 + j]);
      a[j + 8] = f2b(Arow[k0 + 16 + half * 8 + j]);
    }
#pragma unroll
    for (int j = 0; j < 8; ++j) {
      b[j]     = f2b(Brow[k0 + half * 8 + j]);
      b[j + 8] = f2b(Brow[k0 + 16 + half * 8 + j]);
    }
    acc = wmma_bb(a, b, acc);
  }
  if (n < K_) {
#pragma unroll
    for (int e = 0; e < 8; ++e) {
      int m = mt * 16 + e + half * 8;        // C layout: M = e + 8*half
      emi[(size_t)m * K_ + n] = acc[e];
    }
  }
}

// masked-mean softmax entropy of emission
__global__ void k_entropy(const int* amask, char* ws) {
  const float* emi = (const float*)(ws + OFF_EMI);
  float* scal = (float*)(ws + OFF_SCAL);
  int tid = blockIdx.x * 256 + threadIdx.x;
  if (tid >= BT) return;
  const float* row = emi + (size_t)tid * K_;
  float mx = -3e38f;
  for (int k = 0; k < K_; ++k) mx = fmaxf(mx, row[k]);
  float s = 0.f, pl = 0.f;
  for (int k = 0; k < K_; ++k) {
    float e = __expf(row[k] - mx);
    s += e; pl += e * row[k];
  }
  float lse = mx + __logf(s);
  float ent = lse - pl / s;
  float mk = (float)amask[tid];
  atomicAdd(&scal[0], ent * mk);
  atomicAdd(&scal[1], mk);
}

// CRF forward algorithm: sequential in T, one workgroup.
__global__ void k_crf_forward(char* ws) {
  const float* emi = (const float*)(ws + OFF_EMI);
  const float* trg = (const float*)(ws + OFF_TRANS);
  float* alphas = (float*)(ws + OFF_ALPHA);
  __shared__ float tr[K_ * K_];
  __shared__ float ap[B_ * K_];
  int tid = threadIdx.x;
  for (int i = tid; i < K_ * K_; i += 256) tr[i] = trg[i];
  if (tid < B_ * K_) {
    int b = tid / K_, k = tid % K_;
    float a0 = emi[((size_t)b * T_ + 0) * K_ + k];
    ap[tid] = a0;
    alphas[(size_t)(0 * B_ + b) * K_ + k] = a0;
  }
  __syncthreads();
  for (int t = 1; t < T_; ++t) {
    float anew = 0.f;
    if (tid < B_ * K_) {
      int b = tid / K_, k2 = tid % K_;
      float mx = -3e38f;
      for (int k = 0; k < K_; ++k)
        mx = fmaxf(mx, ap[b * K_ + k] + tr[k * K_ + k2]);
      float s = 0.f;
      for (int k = 0; k < K_; ++k)
        s += __expf(ap[b * K_ + k] + tr[k * K_ + k2] - mx);
      anew = mx + __logf(s) + emi[((size_t)b * T_ + t) * K_ + k2];
    }
    __syncthreads();
    if (tid < B_ * K_) {
      ap[tid] = anew;
      alphas[(size_t)(t * B_ + tid / K_) * K_ + (tid % K_)] = anew;
    }
    __syncthreads();
  }
}

// Relaxed backward sampling (Gumbel argmax + softmax relaxation).
__global__ void k_crf_backward(const float* gumbel, char* ws) {
  const float* trg = (const float*)(ws + OFF_TRANS);
  const float* alphas = (const float*)(ws + OFF_ALPHA);
  float* relax = (float*)(ws + OFF_RELAX);
  int* zids = (int*)(ws + OFF_ZIDS);
  __shared__ float tr[K_ * K_];
  __shared__ float sc[B_ * K_];
  __shared__ float yv[B_ * K_];
  __shared__ float lseL[B_], lseY[B_];
  __shared__ int zc[B_];
  int tid = threadIdx.x;
  for (int i = tid; i < K_ * K_; i += 256) tr[i] = trg[i];
  __syncthreads();
  for (int t = T_ - 1; t >= 0; --t) {
    if (tid < B_ * K_) {
      int b = tid / K_, k = tid % K_;
      float a = alphas[(size_t)(t * B_ + b) * K_ + k];
      if (t < T_ - 1) a += tr[k * K_ + zc[b]];
      sc[tid] = a;
    }
    __syncthreads();
    if (tid < B_) {
      float mx = -3e38f;
      for (int k = 0; k < K_; ++k) mx = fmaxf(mx, sc[tid * K_ + k]);
      float s = 0.f;
      for (int k = 0; k < K_; ++k) s += __expf(sc[tid * K_ + k] - mx);
      lseL[tid] = mx + __logf(s);
    }
    __syncthreads();
    if (tid < B_ * K_) {
      int b = tid / K_, k = tid % K_;
      float lg = sc[tid] - lseL[b];
      yv[tid] = lg + gumbel[((size_t)t * B_ + b) * K_ + k];
    }
    __syncthreads();
    if (tid < B_) {
      int best = 0; float bv = yv[tid * K_];
      for (int k = 1; k < K_; ++k) {
        float v = yv[tid * K_ + k];
        if (v > bv) { bv = v; best = k; }
      }
      zc[tid] = best;
      zids[t * B_ + tid] = best;
      float mx = -3e38f;
      for (int k = 0; k < K_; ++k) mx = fmaxf(mx, yv[tid * K_ + k] / TAU);
      float s = 0.f;
      for (int k = 0; k < K_; ++k) s += __expf(yv[tid * K_ + k] / TAU - mx);
      lseY[tid] = mx + __logf(s);
    }
    __syncthreads();
    if (tid < B_ * K_) {
      relax[(size_t)t * B_ * K_ + tid] = __expf(yv[tid] / TAU - lseY[tid / K_]);
    }
    __syncthreads();
  }
}

__global__ void k_writez(char* ws, float* out) {
  int tid = blockIdx.x * 256 + threadIdx.x;
  if (tid < BT) {
    const int* z = (const int*)(ws + OFF_ZIDS);
    int b = tid / T_, t = tid % T_;
    out[1 + tid] = (float)z[t * B_ + b];   // z_sample[b,t]
  }
}

// dec_in[r=t*8+b, h] = relaxed[t,b,:] @ S + embeddings[x[b,t]]  (bf16)
__global__ void k_decin(const float* S, const float* embeddings, const int* x,
                        char* ws) {
  const float* relax = (const float*)(ws + OFF_RELAX);
  __bf16* decin = (__bf16*)(ws + OFF_DECIN);
  size_t idx = (size_t)blockIdx.x * 256 + threadIdx.x;
  if (idx >= (size_t)TM1R * H_) return;
  int h = (int)(idx % H_);
  int r = (int)(idx / H_);
  int b = r % B_, t = r / B_;
  const float* rl = relax + (size_t)r * K_;
  float z = 0.f;
  for (int k = 0; k < K_; ++k) z += rl[k] * S[k * H_ + h];
  int tok = x[b * T_ + t];
  decin[(size_t)r * H_ + h] = f2b(z + embeddings[(size_t)tok * H_ + h]);
}

// xw[2040,3072] = dec_in @ W_ih^T + b_lstm   (WMMA, manual LDS-staged A tile)
__global__ void k_xw(const float* b_lstm, char* ws) {
  const __bf16* decin = (const __bf16*)(ws + OFF_DECIN);
  const __bf16* wih   = (const __bf16*)(ws + OFF_WIH);
  float* xw = (float*)(ws + OFF_XW);
  __shared__ alignas(16) __bf16 atile[16 * AP];
  int mt = blockIdx.x;
  for (int i = threadIdx.x; i < 16 * H_; i += 256) {
    int rr = i / H_, cc = i % H_;
    int r = mt * 16 + rr;
    atile[rr * AP + cc] = (r < TM1R) ? decin[(size_t)r * H_ + cc] : zbf();
  }
  __syncthreads();
  int w = threadIdx.x >> 5;
  int nt = blockIdx.y * 8 + w;               // 192 N tiles
  int lane = threadIdx.x & 31, half = lane >> 4, lm = lane & 15;
  int n = nt * 16 + lm;
  v8f acc = {};
  wmma_k_loop(atile, AP, wih + (size_t)n * H_, acc);
  float bias = b_lstm[n];
#pragma unroll
  for (int e = 0; e < 8; ++e) {
    int r = mt * 16 + e + half * 8;
    if (r < TM1R) xw[(size_t)r * G4H + n] = acc[e] + bias;
  }
}

// One LSTM step: gates[8,3072] = xw[t] + h_prev @ W_hh^T   (WMMA)
__global__ void k_lstm_gemm(int t, char* ws) {
  const __bf16* hbf = (const __bf16*)(ws + OFF_HBF);   // [16,768], rows 8..15 zero
  const __bf16* whh = (const __bf16*)(ws + OFF_WHH);
  const float* xw = (const float*)(ws + OFF_XW);
  float* gates = (float*)(ws + OFF_GATES);
  int w = threadIdx.x >> 5;
  int nt = blockIdx.x * 8 + w;               // 192 N tiles
  int lane = threadIdx.x & 31, half = lane >> 4, lm = lane & 15;
  int n = nt * 16 + lm;
  v8f acc = {};
  wmma_k_loop(hbf, H_, whh + (size_t)n * H_, acc);
#pragma unroll
  for (int e = 0; e < 8; ++e) {
    int m = e + half * 8;
    if (m < B_)
      gates[(size_t)m * G4H + n] = acc[e] + xw[(size_t)(t * B_ + m) * G4H + n];
  }
}

// LSTM pointwise cell update; h written as bf16 (next GEMM + hs buffer).
__global__ void k_lstm_point(int t, char* ws) {
  const float* gates = (const float*)(ws + OFF_GATES);
  float* c = (float*)(ws + OFF_C);
  __bf16* hbf = (__bf16*)(ws + OFF_HBF);
  __bf16* hs = (__bf16*)(ws + OFF_HS);
  int idx = blockIdx.x * 256 + threadIdx.x;
  if (idx >= B_ * H_) return;
  int b = idx / H_, hh = idx % H_;
  float gi = gates[(size_t)b * G4H + hh];
  float gf = gates[(size_t)b * G4H + H_ + hh];
  float gg = gates[(size_t)b * G4H + 2 * H_ + hh];
  float go = gates[(size_t)b * G4H + 3 * H_ + hh];
  float si = 1.f / (1.f + __expf(-gi));
  float sf = 1.f / (1.f + __expf(-gf));
  float so = 1.f / (1.f + __expf(-go));
  float cn = sf * c[idx] + si * tanhf(gg);
  float h = so * tanhf(cn);
  c[idx] = cn;
  hbf[(size_t)b * H_ + hh] = f2b(h);
  hs[(size_t)(t * B_ + b) * H_ + hh] = f2b(h);
}

// Fused logits + online softmax stats per 128-col V chunk. The 16x768 bf16
// A tile is staged into LDS by the Tensor Data Mover: one descriptor-driven
// DMA (iterate mode preserves the banked LDS pitch AP), tracked by TENSORcnt.
__global__ void k_logits(const float* b_out, const int* x, char* ws) {
  const __bf16* hs   = (const __bf16*)(ws + OFF_HS);
  const __bf16* wout = (const __bf16*)(ws + OFF_WOUT);
  float* stats = (float*)(ws + OFF_STATS);
  float* tgtl  = (float*)(ws + OFF_TGTL);
  __shared__ alignas(16) __bf16 atile[16 * AP];
  __shared__ float bm[8][16], bs[8][16];
  int mt = blockIdx.x, ch = blockIdx.y;
#if __has_builtin(__builtin_amdgcn_tensor_load_to_lds) && \
    __has_builtin(__builtin_amdgcn_s_wait_tensorcnt)
  if (threadIdx.x == 0) {
    // D# group 0: count=1 | lds_addr | global_addr[56:0] | type=2
    unsigned ldsoff = (unsigned)(size_t)(void*)atile;      // flat addr[31:0] = LDS byte addr
    unsigned long long ga =
        (unsigned long long)(size_t)hs + (unsigned long long)mt * 16 * H_ * 2;
    u32x4 g0;
    g0[0] = 1u;                                            // count=1, user mode
    g0[1] = ldsoff;
    g0[2] = (unsigned)(ga & 0xFFFFFFFFu);
    g0[3] = (unsigned)((ga >> 32) & 0x01FFFFFFu) | (2u << 30);
    // D# group 1: data_size=2B, iterate_enable, dims/strides in elements
    int rowsLeft = TM1R - mt * 16;                          // OOB rows read 0
    if (rowsLeft < 0) rowsLeft = 0;
    i32x8 g1;
    g1[0] = (1 << 16) | (1 << 19);                          // data_size=1(2B), iterate
    g1[1] = (int)((H_ & 0xFFFF) << 16);                     // tensor_dim0 lo16 @bit48
    g1[2] = (int)((H_ >> 16) & 0xFFFF) |                    // tensor_dim0 hi16
            (int)((rowsLeft & 0xFFFF) << 16);               // tensor_dim1 lo16 @bit80
    g1[3] = (int)((rowsLeft >> 16) & 0xFFFF) |              // tensor_dim1 hi16
            (int)(H_ << 16);                                // tile_dim0 @bit112
    g1[4] = 1;                                              // tile_dim1=1 (one row/iter)
    g1[5] = H_;                                             // tensor_dim0_stride lo32
    g1[6] = 0;
    g1[7] = 0;
    // D# group 2 (iterate mode): lds_addr_increment @63:32,
    // global_addr_increment @111:64, iterate_count @127:112 (15 -> 16 iters)
    i32x4 g2;
    g2[0] = 0;
    g2[1] = AP;                                             // LDS pitch per iter
    g2[2] = H_;                                             // one tensor row per iter
    g2[3] = (15 << 16);
    i32x4 g3 = {0, 0, 0, 0};
    i32x8 g4 = {0, 0, 0, 0, 0, 0, 0, 0};                    // trailing group (clang-23 form)
    __builtin_amdgcn_tensor_load_to_lds(g0, g1, g2, g3, g4, 0);
  }
  __builtin_amdgcn_s_wait_tensorcnt(0);
#else
  for (int i = threadIdx.x; i < 16 * H_; i += 256) {
    int rr = i / H_, cc = i % H_;
    int r = mt * 16 + rr;
    atile[rr * AP + cc] = (r < TM1R) ? hs[(size_t)r * H_ + cc] : zbf();
  }
#endif
  __syncthreads();
  int w = threadIdx.x >> 5;
  int lane = threadIdx.x & 31, half = lane >> 4, lm = lane & 15;
  int n = ch * 128 + w * 16 + lm;            // < VPAD, padded W_out rows are 0
  v8f acc = {};
  wmma_k_loop(atile, AP, wout + (size_t)n * H_, acc);
  bool nv = n < V_;
  float bias = nv ? b_out[n] : 0.f;
#pragma unroll
  for (int e = 0; e < 8; ++e) {
    int r = mt * 16 + e + half * 8;
    bool rv = r < TM1R;
    float lg = (nv && rv) ? acc[e] + bias : -3e38f;
    if (nv && rv) {
      int b = r & 7, tt = r >> 3;
      if (n == x[b * T_ + tt + 1]) tgtl[r] = lg;   // unique (row,col) hit
    }
    // per-row (max,sumexp) across the 16 lanes of this half-wave
    float m = lg, s = (lg > -1e37f) ? 1.f : 0.f;
#pragma unroll
    for (int d = 1; d < 16; d <<= 1) {
      float mo = __shfl_xor(m, d, 32);
      float so = __shfl_xor(s, d, 32);
      float m2 = fmaxf(m, mo);
      s = s * __expf(m - m2) + so * __expf(mo - m2);
      m = m2;
    }
    if (lm == 0) { bm[w][e + half * 8] = m; bs[w][e + half * 8] = s; }
  }
  __syncthreads();
  if (threadIdx.x < 16) {
    int rr = threadIdx.x;
    float m = -3e38f, s = 0.f;
    for (int ww = 0; ww < 8; ++ww) {
      float mo = bm[ww][rr], so = bs[ww][rr];
      float m2 = fmaxf(m, mo);
      s = s * __expf(m - m2) + so * __expf(mo - m2);
      m = m2;
    }
    int r = mt * 16 + rr;
    if (r < TM1R) {
      stats[((size_t)r * NCHUNK + ch) * 2 + 0] = m;
      stats[((size_t)r * NCHUNK + ch) * 2 + 1] = s;
    }
  }
}

// Merge chunk stats -> logsumexp; masked mean token logprob; loss.
__global__ void k_final(const int* amask, char* ws, float* out) {
  const float* stats = (const float*)(ws + OFF_STATS);
  const float* tgtl  = (const float*)(ws + OFF_TGTL);
  const float* scal  = (const float*)(ws + OFF_SCAL);
  __shared__ float ssum[256], scnt[256];
  float as = 0.f, ac = 0.f;
  for (int r = threadIdx.x; r < TM1R; r += 256) {
    float m = -3e38f, s = 0.f;
    for (int ch = 0; ch < NCHUNK; ++ch) {
      float mo = stats[((size_t)r * NCHUNK + ch) * 2 + 0];
      float so = stats[((size_t)r * NCHUNK + ch) * 2 + 1];
      float m2 = fmaxf(m, mo);
      s = s * __expf(m - m2) + so * __expf(mo - m2);
      m = m2;
    }
    float lse = m + __logf(s);
    float lp = tgtl[r] - lse;
    int b = r & 7, t = r >> 3;
    float mk = (float)amask[b * T_ + t + 1];
    as += lp * mk; ac += mk;
  }
  ssum[threadIdx.x] = as; scnt[threadIdx.x] = ac;
  __syncthreads();
  for (int st = 128; st > 0; st >>= 1) {
    if (threadIdx.x < st) {
      ssum[threadIdx.x] += ssum[threadIdx.x + st];
      scnt[threadIdx.x] += scnt[threadIdx.x + st];
    }
    __syncthreads();
  }
  if (threadIdx.x == 0) {
    float p = ssum[0] / scnt[0];
    float ent = scal[0] / scal[1];
    out[0] = -(p + Z_BETA * ent);
  }
}

// ---------------- launch ----------------
extern "C" void kernel_launch(void* const* d_in, const int* in_sizes, int n_in,
                              void* d_out, int out_size, void* d_ws,
                              size_t ws_size, hipStream_t stream) {
  (void)in_sizes; (void)n_in; (void)out_size; (void)ws_size;
  const float* x_emb = (const float*)d_in[0];
  const float* S     = (const float*)d_in[1];
  const float* emb   = (const float*)d_in[2];
  const float* Wih   = (const float*)d_in[3];
  const float* Whh   = (const float*)d_in[4];
  const float* bl    = (const float*)d_in[5];
  const float* Wout  = (const float*)d_in[6];
  const float* bo    = (const float*)d_in[7];
  const float* gum   = (const float*)d_in[8];
  const int*   x     = (const int*)d_in[9];
  const int*   am    = (const int*)d_in[10];
  char* ws = (char*)d_ws;
  float* out = (float*)d_out;

  size_t nconv = (size_t)G4H * H_ * 2 + (size_t)VPAD * H_ +
                 (size_t)16 * H_ + (size_t)B_ * H_ + 16;
  k_convert<<<(unsigned)((nconv + 255) / 256), 256, 0, stream>>>(Wih, Whh, Wout, ws);
  k_transition<<<1, 512, 0, stream>>>(S, ws);
  k_emission<<<32, 256, 0, stream>>>(x_emb, S, ws);
  k_entropy<<<BT / 256, 256, 0, stream>>>(am, ws);
  k_crf_forward<<<1, 256, 0, stream>>>(ws);
  k_crf_backward<<<1, 256, 0, stream>>>(gum, ws);
  k_writez<<<BT / 256, 256, 0, stream>>>(ws, out);
  {
    size_t tot = (size_t)TM1R * H_;
    k_decin<<<(unsigned)((tot + 255) / 256), 256, 0, stream>>>(S, emb, x, ws);
  }
  k_xw<<<dim3(128, 24), 256, 0, stream>>>(bl, ws);
  for (int t = 0; t < T_ - 1; ++t) {
    k_lstm_gemm<<<24, 256, 0, stream>>>(t, ws);
    k_lstm_point<<<24, 256, 0, stream>>>(t, ws);
  }
  k_logits<<<dim3(128, NCHUNK), 256, 0, stream>>>(bo, x, ws);
  k_final<<<1, 256, 0, stream>>>(am, ws, out);
}
